// MoEHead_2877628088602
// MI455X (gfx1250) — matrix operand order repaired and checked by reading, not compiled, not run on hardware
//
#include <hip/hip_runtime.h>
#include <math.h>

// MoE head for MI455X (gfx1250): wave32, split-bf16 WMMA, TDM async staging.
// d_in: x (T*DH f32), Wr (NE*DH f32), W_in (NE*DHID*DH f32), W_out (NE*DHID*DH f32)
// d_out: y (T*DH f32)

#define T_TOK 4096
#define DH 512
#define DHID 2048
#define NE 8

#define ROWS 32       // gathered token rows per block
#define CJ 32         // DHID chunk per iteration (== bf16 WMMA K)
#define NJC (DHID / CJ)
#define XS 520        // LDS stride (halves) for 512-wide rows (pad 4 dwords/256)
#define OS 40         // LDS stride (halves) for 32-wide rows  (pad 4 dwords/16)

typedef float v8f __attribute__((ext_vector_type(8)));
typedef __bf16 v16bf __attribute__((ext_vector_type(16)));
typedef unsigned int v4u __attribute__((ext_vector_type(4)));
typedef int v8i_ __attribute__((ext_vector_type(8)));
typedef int v4i_ __attribute__((ext_vector_type(4)));

__device__ __forceinline__ v8f wmma_bf16(v16bf a, v16bf b, v8f c) {
  // D(16x16 f32) = A(16x32 bf16) * B(32x16 bf16) + C
  return __builtin_amdgcn_wmma_f32_16x16x32_bf16(false, a, false, b, (short)0,
                                                 c, false, false);
}

// A-fragment K permutation within each 32-group: pos bits p3=k4, p4=k3
__device__ __forceinline__ int permA(int k) {
  return (k & 7) | ((k >> 1) & 8) | ((k << 1) & 16);
}

// ---------------------------------------------------------------------------
// TDM: 2-D bf16 tile load global->LDS with hardware LDS row padding.
// D# packing per CDNA5 ISA 8.3/8.4 (count=1, type=2 "image", data_size=2B).
// amdgpu-toolchain 6-arg builtin form: (v4u, v8i, v4i, v4i, v8i, i32 cpol).
// ---------------------------------------------------------------------------
__device__ __forceinline__ void tdm_load_bf16(
    unsigned int lds_byte_off, const __bf16* gptr, unsigned int tile_d0,
    unsigned int tile_d1, unsigned int tensor_d0, unsigned int tensor_d1,
    unsigned long long stride_d0, int pad_interval_code, int pad_amount_code) {
  unsigned long long ga = (unsigned long long)(const void*)gptr;
  v4u g0;
  g0[0] = 1u;                                // count=1, user descriptor
  g0[1] = lds_byte_off;                      // lds_addr
  g0[2] = (unsigned int)ga;                  // global_addr[31:0]
  g0[3] = ((unsigned int)(ga >> 32) & 0x01FFFFFFu) | (2u << 30);  // type=2
  v8i_ g1;
  g1[0] = (int)((1u << 16)                   // data_size = 2 bytes
                | (1u << 20)                 // pad_enable
                | ((unsigned int)pad_interval_code << 22)
                | ((unsigned int)pad_amount_code << 25));
  g1[1] = (int)((tensor_d0 & 0xFFFFu) << 16);
  g1[2] = (int)(((tensor_d0 >> 16) & 0xFFFFu) | ((tensor_d1 & 0xFFFFu) << 16));
  g1[3] = (int)(((tensor_d1 >> 16) & 0xFFFFu) | ((tile_d0 & 0xFFFFu) << 16));
  g1[4] = (int)(tile_d1 & 0xFFFFu);          // tile_dim1; tile_dim2 = 0
  g1[5] = (int)(unsigned int)(stride_d0 & 0xFFFFFFFFull);
  g1[6] = (int)(unsigned int)((stride_d0 >> 32) & 0xFFFFull);
  g1[7] = 0;
  v4i_ z4 = {0, 0, 0, 0};                    // D# groups 2/3: unused (2-D)
  v8i_ z8 = {0, 0, 0, 0, 0, 0, 0, 0};
  __builtin_amdgcn_tensor_load_to_lds(g0, g1, z4, z4, z8, 0);
}

// ---------------------------------------------------------------------------
// Pre-pass A: split f32 -> bf16 hi/lo planes, rows of 512, natural K order
// (no permutation: output index = input index) (W_in).
// ---------------------------------------------------------------------------
__global__ __launch_bounds__(256) void cvt_split_perm(
    const float* __restrict__ src, __bf16* __restrict__ dh,
    __bf16* __restrict__ dl, int n) {
  int i = blockIdx.x * 256 + threadIdx.x;
  if (i >= n) return;
  float v = src[i];
  __bf16 hi = (__bf16)v;
  __bf16 lo = (__bf16)(v - (float)hi);
  int o = i;
  dh[o] = hi;
  dl[o] = lo;
}

// ---------------------------------------------------------------------------
// Pre-pass B: W_out -> transposed bf16 hi/lo planes [e][n=DH][j=DHID],
// natural K order (B operand). LDS-tiled 32x32 transpose.
// ---------------------------------------------------------------------------
__global__ __launch_bounds__(256) void cvt_wout_t(
    const float* __restrict__ src, __bf16* __restrict__ dh,
    __bf16* __restrict__ dl) {
  __shared__ float TILE[32 * 33];
  const int e = blockIdx.z;
  const int j0 = blockIdx.x * 32;
  const int n0 = blockIdx.y * 32;
  const int a = threadIdx.x >> 5;
  const int b = threadIdx.x & 31;
#pragma unroll
  for (int p = 0; p < 4; ++p) {
    int j = a + p * 8;
    TILE[j * 33 + b] = src[((size_t)e * DHID + (j0 + j)) * DH + n0 + b];
  }
  __syncthreads();
#pragma unroll
  for (int p = 0; p < 4; ++p) {
    int n = a + p * 8;
    float v = TILE[b * 33 + n];  // = W_out[e][j0+b][n0+n]
    __bf16 hi = (__bf16)v;
    size_t o = ((size_t)e * DH + (n0 + n)) * DHID + j0 + b;
    dh[o] = hi;
    dl[o] = (__bf16)(v - (float)hi);
  }
}

// ---------------------------------------------------------------------------
// Kernel 1: router (fp32 VALU, exact top-k). One wave32 per token.
// ---------------------------------------------------------------------------
__global__ __launch_bounds__(256) void moe_router(
    const float* __restrict__ x, const float* __restrict__ Wr,
    int* __restrict__ cnt, int* __restrict__ gidx, float* __restrict__ gw) {
  const int lane = threadIdx.x & 31;
  const int wave = threadIdx.x >> 5;
  const int t = blockIdx.x * 8 + wave;
  if (t >= T_TOK) return;

  float acc[NE];
#pragma unroll
  for (int e = 0; e < NE; ++e) acc[e] = 0.f;
  for (int d = lane; d < DH; d += 32) {
    float xv = x[(size_t)t * DH + d];
#pragma unroll
    for (int e = 0; e < NE; ++e) acc[e] += xv * Wr[e * DH + d];
  }
#pragma unroll
  for (int e = 0; e < NE; ++e) {
#pragma unroll
    for (int off = 16; off > 0; off >>= 1) acc[e] += __shfl_xor(acc[e], off, 32);
  }
  if (lane == 0) {
    int e1 = 0;
    float v1 = acc[0];
#pragma unroll
    for (int e = 1; e < NE; ++e)
      if (acc[e] > v1) { v1 = acc[e]; e1 = e; }
    int e2 = -1;
    float v2 = -3.402823466e38f;
#pragma unroll
    for (int e = 0; e < NE; ++e)
      if (e != e1 && acc[e] > v2) { v2 = acc[e]; e2 = e; }
    float p = __expf(v2 - v1);
    float inv = 1.f / (1.f + p);
    int s1 = atomicAdd(&cnt[e1], 1);
    gidx[e1 * T_TOK + s1] = t;
    gw[e1 * T_TOK + s1] = inv;
    int s2 = atomicAdd(&cnt[e2], 1);
    gidx[e2 * T_TOK + s2] = t;
    gw[e2 * T_TOK + s2] = p * inv;
  }
}

// ---------------------------------------------------------------------------
// Kernel 2: fused expert FFN over gathered rows, split-bf16 WMMA + TDM
// double-buffered weight staging. grid = (T/ROWS, NE), 256 threads = 8 waves.
// ---------------------------------------------------------------------------
__global__ __launch_bounds__(256) void moe_expert(
    const float* __restrict__ x, const __bf16* __restrict__ WIHg,
    const __bf16* __restrict__ WILg, const __bf16* __restrict__ WOTHg,
    const __bf16* __restrict__ WOTLg, const int* __restrict__ cnt,
    const int* __restrict__ gidx, const float* __restrict__ gw,
    float* __restrict__ out) {
  const int e = blockIdx.y;
  const int nrows = cnt[e];
  const int rbase = blockIdx.x * ROWS;
  if (rbase >= nrows) return;

  __shared__ __attribute__((aligned(16))) __bf16 XGH[ROWS * XS];
  __shared__ __attribute__((aligned(16))) __bf16 XGL[ROWS * XS];
  __shared__ __attribute__((aligned(16))) __bf16 WIH2[2][CJ * XS];
  __shared__ __attribute__((aligned(16))) __bf16 WIL2[2][CJ * XS];
  __shared__ __attribute__((aligned(16))) __bf16 WOH[DH * OS];
  __shared__ __attribute__((aligned(16))) __bf16 WOL[DH * OS];
  __shared__ __attribute__((aligned(16))) float HA[CJ * ROWS];
  __shared__ __attribute__((aligned(16))) float HB[CJ * ROWS];
  __shared__ __attribute__((aligned(16))) __bf16 HCH[ROWS * OS];
  __shared__ __attribute__((aligned(16))) __bf16 HCL[ROWS * OS];
  __shared__ int TOK[ROWS];
  __shared__ float WRW[ROWS];

  const int tid = threadIdx.x;
  const int lane = tid & 31;
  const int wave = tid >> 5;
  const int lrow = lane & 15;
  const int lhalf = lane >> 4;

  if (tid < ROWS) {
    int slot = rbase + tid;
    int tok = -1;
    float wv = 0.f;
    if (slot < nrows) {
      tok = gidx[e * T_TOK + slot];
      wv = gw[e * T_TOK + slot];
    }
    TOK[tid] = tok;
    WRW[tid] = wv;
  }

  // Kick off WI chunk 0 via TDM (wave 0 owns all tensor ops)
  const __bf16* WIHe = WIHg + (size_t)e * DHID * DH;
  const __bf16* WILe = WILg + (size_t)e * DHID * DH;
  const __bf16* WOHe = WOTHg + (size_t)e * DH * DHID;
  const __bf16* WOLe = WOTLg + (size_t)e * DH * DHID;
  if (wave == 0) {
    // W_in chunk: 2D tile 512 x 32 rows, contiguous; LDS rows padded to 520
    tdm_load_bf16((unsigned int)(size_t)(void*)WIH2[0], WIHe, DH, CJ, DH, CJ,
                  DH, /*pad: 256dw*/ 7, /*4dw*/ 3);
    tdm_load_bf16((unsigned int)(size_t)(void*)WIL2[0], WILe, DH, CJ, DH, CJ,
                  DH, 7, 3);
  }
  __syncthreads();

  // stage gathered x rows as bf16 hi/lo planes (B operand: natural K order)
  for (int i = tid; i < ROWS * DH; i += 256) {
    int r = i >> 9;
    int d = i & (DH - 1);
    int tok = TOK[r];
    float v = (tok >= 0) ? x[(size_t)tok * DH + d] : 0.f;
    __bf16 hi = (__bf16)v;
    XGH[r * XS + d] = hi;
    XGL[r * XS + d] = (__bf16)(v - (float)hi);
  }

  // GEMM1 role: 4 tiles of H^T (2 j-tiles x 2 r-tiles), K split in halves
  const int tset = wave & 3;
  const int khalf = wave >> 2;
  const int g1_j0 = (tset >> 1) * 16;
  const int g1_r0 = (tset & 1) * 16;
  // GEMM2 role: wave owns rows [g2_r0,+16), cols [g2_n0,+128)
  const int g2_r0 = (wave >> 2) * 16;
  const int g2_n0 = (wave & 3) * 128;

  v8f yacc[8];
#pragma unroll
  for (int c = 0; c < 8; ++c) {
#pragma unroll
    for (int v = 0; v < 8; ++v) yacc[c][v] = 0.f;
  }

  for (int jc = 0; jc < NJC; ++jc) {
    __syncthreads();  // all waves done with prev WOT/WI[jc&1^1]/HC
    if (wave == 0) {
      // W_out^T chunk: 2D tile 32 x 512 rows, row stride DHID; LDS rows -> 40
      tdm_load_bf16((unsigned int)(size_t)(void*)WOH, WOHe + jc * CJ, CJ, DH,
                    DHID, DH, DHID, /*pad: 16dw*/ 3, /*4dw*/ 3);
      tdm_load_bf16((unsigned int)(size_t)(void*)WOL, WOLe + jc * CJ, CJ, DH,
                    DHID, DH, DHID, 3, 3);
      if (jc + 1 < NJC) {  // prefetch next W_in chunk into other buffer
        const size_t co = (size_t)(jc + 1) * CJ * DH;
        tdm_load_bf16((unsigned int)(size_t)(void*)WIH2[(jc + 1) & 1],
                      WIHe + co, DH, CJ, DH, CJ, DH, 7, 3);
        tdm_load_bf16((unsigned int)(size_t)(void*)WIL2[(jc + 1) & 1],
                      WILe + co, DH, CJ, DH, CJ, DH, 7, 3);
        __builtin_amdgcn_s_wait_tensorcnt(4);  // WI[jc] pair retired
      } else {
        __builtin_amdgcn_s_wait_tensorcnt(2);
      }
    }
    __syncthreads();  // WI[jc] visible to all waves

    // --- GEMM1: H^T[j][r] = sum_d Win[j][d]*Xg[r][d], split-bf16, K half ---
    {
      const __bf16* WIHc = WIH2[jc & 1];
      const __bf16* WILc = WIL2[jc & 1];
      v8f h;
#pragma unroll
      for (int v = 0; v < 8; ++v) h[v] = 0.f;
      const int arow = (g1_j0 + lrow) * XS;
      const int brow = (g1_r0 + lrow) * XS;
#pragma unroll
      for (int kk = 0; kk < 8; ++kk) {
        int off = (khalf * 8 + kk) * 32 + 16 * lhalf;
        v16bf ah = *(const v16bf*)&WIHc[arow + off];
        v16bf al = *(const v16bf*)&WILc[arow + off];
        v16bf bh = *(const v16bf*)&XGH[brow + off];
        v16bf bl = *(const v16bf*)&XGL[brow + off];
        h = wmma_bf16(ah, bh, h);
        h = wmma_bf16(ah, bl, h);
        h = wmma_bf16(al, bh, h);
      }
      float* H = khalf ? HB : HA;
#pragma unroll
      for (int v = 0; v < 8; ++v)
        H[(g1_j0 + v + 8 * lhalf) * ROWS + (g1_r0 + lrow)] = h[v];
    }
    __syncthreads();

    // --- combine K-halves, exact (erf) gelu, routing weight, split bf16 ---
    for (int i = tid; i < CJ * ROWS; i += 256) {
      int j = i >> 5;
      int r = i & (ROWS - 1);
      float s = HA[i] + HB[i];
      float g = 0.5f * s * (1.f + erff(s * 0.70710678118654752f));
      float hv = g * WRW[r];
      __bf16 hi = (__bf16)hv;
      int o = r * OS + j;
      HCH[o] = hi;
      HCL[o] = (__bf16)(hv - (float)hi);
    }
    if (wave == 0) {  // WOT[jc] DMA complete (only WI prefetch may remain)
      if (jc + 1 < NJC) {
        __builtin_amdgcn_s_wait_tensorcnt(2);
      } else {
        __builtin_amdgcn_s_wait_tensorcnt(0);
      }
    }
    __syncthreads();

    // --- GEMM2: yacc += Hc(32 x 32) * Wout_chunk(32 x 512), K=32 ---
    {
      v16bf ah = *(const v16bf*)&HCH[(g2_r0 + lrow) * OS + 16 * lhalf];
      v16bf al = *(const v16bf*)&HCL[(g2_r0 + lrow) * OS + 16 * lhalf];
#pragma unroll
      for (int c = 0; c < 8; ++c) {
        int n = g2_n0 + c * 16 + lrow;
        v16bf bh = *(const v16bf*)&WOH[n * OS + 16 * lhalf];
        v16bf bl = *(const v16bf*)&WOL[n * OS + 16 * lhalf];
        yacc[c] = wmma_bf16(ah, bh, yacc[c]);
        yacc[c] = wmma_bf16(ah, bl, yacc[c]);
        yacc[c] = wmma_bf16(al, bh, yacc[c]);
      }
    }
  }

  // --- scatter-add output tile (two experts per token commute exactly) ---
#pragma unroll
  for (int c = 0; c < 8; ++c) {
    int n = g2_n0 + c * 16 + lrow;
#pragma unroll
    for (int v = 0; v < 8; ++v) {
      int r = g2_r0 + v + 8 * lhalf;
      int tok = TOK[r];
      if (tok >= 0) atomicAdd(out + (size_t)tok * DH + n, yacc[c][v]);
    }
  }
}

extern "C" void kernel_launch(void* const* d_in, const int* in_sizes, int n_in,
                              void* d_out, int out_size, void* d_ws,
                              size_t ws_size, hipStream_t stream) {
  const float* x = (const float*)d_in[0];
  const float* Wr = (const float*)d_in[1];
  const float* Win = (const float*)d_in[2];
  const float* Wout = (const float*)d_in[3];
  float* out = (float*)d_out;

  char* ws = (char*)d_ws;
  int* cnt = (int*)ws;                                      // NE ints
  int* gidx = (int*)(ws + 256);                             // NE*T ints
  float* gw = (float*)(ws + 256 + (size_t)NE * T_TOK * 4);  // NE*T floats

  const size_t NW = (size_t)NE * DHID * DH;  // elements per plane
  __bf16* WIH = (__bf16*)(ws + (1 << 19));
  __bf16* WIL = WIH + NW;
  __bf16* WOTH = WIL + NW;
  __bf16* WOTL = WOTH + NW;

  (void)hipMemsetAsync(cnt, 0, NE * sizeof(int), stream);
  (void)hipMemsetAsync(d_out, 0, (size_t)T_TOK * DH * sizeof(float), stream);

  cvt_split_perm<<<(int)(NW / 256), 256, 0, stream>>>(Win, WIH, WIL, (int)NW);
  cvt_wout_t<<<dim3(DHID / 32, DH / 32, NE), 256, 0, stream>>>(Wout, WOTH,
                                                               WOTL);
  moe_router<<<T_TOK / 8, 256, 0, stream>>>(x, Wr, cnt, gidx, gw);
  moe_expert<<<dim3(T_TOK / ROWS, NE), 256, 0, stream>>>(
      x, WIH, WIL, WOTH, WOTL, cnt, gidx, gw, out);
}
